// MultiheadCrossAttention_22686017257525
// MI455X (gfx1250) — matrix operand order repaired and checked by reading, not compiled, not run on hardware
//
#include <hip/hip_runtime.h>
#include <hip/hip_bf16.h>

// ---------------------------------------------------------------------------
// MI455X / gfx1250 multi-head cross attention.
// All GEMMs on v_wmma_f32_16x16x32_f16 (fp32 accumulate); flash-style softmax;
// global->LDS staging through GLOBAL_LOAD_ASYNC_TO_LDS_B128 when available.
// ---------------------------------------------------------------------------

typedef __attribute__((ext_vector_type(16))) _Float16 v16h;
typedef __attribute__((ext_vector_type(8)))  float    v8f;

union FragH { v16h v; unsigned int u[8]; _Float16 h[16]; };
union U32H  { unsigned int u; _Float16 h[2]; };

#define WMMA_F16(a, b, c) \
  __builtin_amdgcn_wmma_f32_16x16x32_f16(false, (a), false, (b), (short)0, (c), false, false)

#if __has_builtin(__builtin_amdgcn_global_load_async_to_lds_b128) && \
    __has_builtin(__builtin_amdgcn_s_wait_asynccnt)
#define HAVE_ASYNC_LDS 1
// Exact parameter types per hipcc diagnostic:
//   param0: __attribute__((vector_size(16))) int  __device__ * (AS1, non-const)
//   param1: same vector type in LDS (AS3); then imm offset, imm cpol.
typedef int vint4 __attribute__((__vector_size__(16)));
typedef __attribute__((address_space(1))) vint4 as1_v4i;
typedef __attribute__((address_space(3))) vint4 as3_v4i;
#define ASYNC_LDS_B128(gsrc, ldst)                            \
  __builtin_amdgcn_global_load_async_to_lds_b128(             \
      (as1_v4i*)(gsrc), (as3_v4i*)(ldst), 0, 0)
#else
#define HAVE_ASYNC_LDS 0
#endif

// ---------------------------------------------------------------------------
// fp32 -> f16 conversion (grid-stride)
// ---------------------------------------------------------------------------
__global__ void cvt_f32_to_f16(const float* __restrict__ in,
                               _Float16* __restrict__ out, int n) {
  int i = blockIdx.x * blockDim.x + threadIdx.x;
  int stride = gridDim.x * blockDim.x;
  for (; i < n; i += stride) out[i] = (_Float16)in[i];
}

// ---------------------------------------------------------------------------
// Tiled WMMA GEMM:  Out[M,N] = A[M,K] * W[K,N] + bias[N]
// 128 threads (4 waves); 64x64 tile per block, 16x64 per wave.
// A staged row-major (async b128 copies when available); W staged transposed.
// ---------------------------------------------------------------------------
template <bool OUT_F32>
__global__ __launch_bounds__(128) void gemm_wmma(
    const _Float16* __restrict__ A, const _Float16* __restrict__ W,
    const float* __restrict__ bias, void* __restrict__ outp,
    int M, int K, int N) {
  __shared__ _Float16 sA[64][32];   // [m][k]
  __shared__ _Float16 sWt[64][32];  // [n][k]  (transposed W tile)

  const int tid  = threadIdx.x;
  const int lane = tid & 31;
  const int wave = tid >> 5;
  const int kh   = (lane >> 4) & 1;
  const int ln   = lane & 15;
  const int row0 = blockIdx.x * 64;
  const int col0 = blockIdx.y * 64;

  const v8f zero = {0.f, 0.f, 0.f, 0.f, 0.f, 0.f, 0.f, 0.f};
  v8f acc[4];
#pragma unroll
  for (int t = 0; t < 4; ++t) acc[t] = zero;

  const int kIters = K >> 5;
  for (int kk = 0; kk < kIters; ++kk) {
    const int k0 = kk * 32;
    __syncthreads();
    // Stage A tile: 64x32 halves = 256 uint4, 2 per thread.
#pragma unroll
    for (int j = 0; j < 2; ++j) {
      int idx = tid + j * 128;
      int r = idx >> 2, c4 = idx & 3;
      const uint4* src =
          reinterpret_cast<const uint4*>(A + (size_t)(row0 + r) * K + k0);
#if HAVE_ASYNC_LDS
      ASYNC_LDS_B128(src + c4, &sA[r][c4 * 8]);
#else
      *reinterpret_cast<uint4*>(&sA[r][c4 * 8]) = src[c4];
#endif
    }
    // Stage W tile transposed: 32x64 halves = 1024 uints, 8 per thread.
#pragma unroll
    for (int j = 0; j < 8; ++j) {
      int idx = tid + j * 128;
      int kr = idx >> 5, cu = idx & 31;
      U32H w;
      w.u = *reinterpret_cast<const unsigned int*>(
          W + (size_t)(k0 + kr) * N + col0 + cu * 2);
      sWt[cu * 2 + 0][kr] = w.h[0];
      sWt[cu * 2 + 1][kr] = w.h[1];
    }
#if HAVE_ASYNC_LDS
    __builtin_amdgcn_s_wait_asynccnt(0);
#endif
    __syncthreads();

    // A fragment: rows wave*16 .. +15, K-pattern per ISA table.
    FragH fa;
#pragma unroll
    for (int j = 0; j < 8; ++j) {
      const int kA = 2 * j + ((j >= 4) ? 8 : 0) + 8 * kh;
      fa.u[j] =
          *reinterpret_cast<const unsigned int*>(&sA[wave * 16 + ln][kA]);
    }
#pragma unroll
    for (int t = 0; t < 4; ++t) {
      FragH fb;
#pragma unroll
      for (int j = 0; j < 8; ++j)
        fb.u[j] = *reinterpret_cast<const unsigned int*>(
            &sWt[t * 16 + ln][2 * j + ((j >= 4) ? 8 : 0) + 8 * kh]);
      acc[t] = WMMA_F16(fa.v, fb.v, acc[t]);
    }
  }

  // Epilogue: bias add + store (fp32 or f16).
#pragma unroll
  for (int t = 0; t < 4; ++t) {
    const int n = col0 + t * 16 + ln;
    const float bv = bias[n];
#pragma unroll
    for (int r = 0; r < 8; ++r) {
      const int m = row0 + wave * 16 + r + 8 * kh;
      const float v = acc[t][r] + bv;
      if (OUT_F32)
        reinterpret_cast<float*>(outp)[(size_t)m * N + n] = v;
      else
        reinterpret_cast<_Float16*>(outp)[(size_t)m * N + n] = (_Float16)v;
    }
  }
}

// ---------------------------------------------------------------------------
// Flash attention: one (batch, head, 64-row q block) per workgroup.
// 4 waves x 16 q-rows. Online softmax with scaling = 1/384 (emb_dim/2).
// Q/K/V are (B*1024, 768) f16; head h occupies columns [h*64, h*64+64).
// Softmax statistics use a row-per-lane LDS layout: lane (ln, half) owns
// 16 columns of row ln -> one shfl_xor(16) per reduction instead of a
// 4-step bpermute tree per row.
// ---------------------------------------------------------------------------
__global__ __launch_bounds__(128) void attn_wmma(
    const _Float16* __restrict__ Q, const _Float16* __restrict__ Kb,
    const _Float16* __restrict__ Vb, _Float16* __restrict__ O) {
  __shared__ _Float16 sK[32][64];     // [k_local][d]   (row-major for Q*K^T)
  __shared__ _Float16 sVt[64][32];    // [d][k_local]   (transposed for P*V)
  __shared__ _Float16 sP[4][16][32];  // per-wave P tile (f16, A-frag source)
  __shared__ float    sS[4][16][32];  // per-wave raw scores (fp32)

  const int tid  = threadIdx.x;
  const int lane = tid & 31;
  const int wave = tid >> 5;
  const int kh   = (lane >> 4) & 1;
  const int ln   = lane & 15;
  const int b    = blockIdx.z;
  const int h    = blockIdx.y;
  const int q0   = blockIdx.x * 64;

  // Load this wave's Q tile (16 rows x 64 dims) as two A-fragments.
  const size_t baseQ =
      ((size_t)b * 1024 + q0 + wave * 16 + ln) * 768 + (size_t)h * 64;
  FragH aq[2];
#pragma unroll
  for (int c = 0; c < 2; ++c)
#pragma unroll
    for (int j = 0; j < 8; ++j) {
      const int kA = 2 * j + ((j >= 4) ? 8 : 0) + 8 * kh;
      aq[c].u[j] =
          *reinterpret_cast<const unsigned int*>(Q + baseQ + c * 32 + kA);
    }

  const v8f zero = {0.f, 0.f, 0.f, 0.f, 0.f, 0.f, 0.f, 0.f};
  v8f accO[4];
#pragma unroll
  for (int t = 0; t < 4; ++t) accO[t] = zero;
  float rm = -3.0e38f;  // running max of row `ln` (row-per-lane layout)
  float rs = 0.f;       // running sum of row `ln`

  const float scale = 1.0f / 384.0f;  // emb_dim ** 1 / 2 == emb_dim / 2

  for (int kb = 0; kb < 32; ++kb) {
    __syncthreads();
    // Stage K tile (32x64) row-major: 256 x b128, 2 per thread.
#pragma unroll
    for (int j = 0; j < 2; ++j) {
      int idx = tid + j * 128;
      int r = idx >> 3, c4 = idx & 7;
      const uint4* src = reinterpret_cast<const uint4*>(
          Kb + ((size_t)b * 1024 + kb * 32 + r) * 768 + (size_t)h * 64);
#if HAVE_ASYNC_LDS
      ASYNC_LDS_B128(src + c4, &sK[r][c4 * 8]);
#else
      *reinterpret_cast<uint4*>(&sK[r][c4 * 8]) = src[c4];
#endif
    }
    // Stage V tile transposed: 1024 uints, 8 per thread.
#pragma unroll
    for (int j = 0; j < 8; ++j) {
      int idx = tid + j * 128;
      int r = idx >> 5, cu = idx & 31;
      U32H w;
      w.u = *reinterpret_cast<const unsigned int*>(
          Vb + ((size_t)b * 1024 + kb * 32 + r) * 768 + (size_t)h * 64 +
          cu * 2);
      sVt[cu * 2 + 0][r] = w.h[0];
      sVt[cu * 2 + 1][r] = w.h[1];
    }
#if HAVE_ASYNC_LDS
    __builtin_amdgcn_s_wait_asynccnt(0);
#endif
    __syncthreads();

    // S = (Q * K^T) * scale, two 16x16 tiles covering 32 keys.
    v8f sc[2];
#pragma unroll
    for (int nt = 0; nt < 2; ++nt) {
      v8f s = zero;
#pragma unroll
      for (int c = 0; c < 2; ++c) {
        FragH fb;
#pragma unroll
        for (int j = 0; j < 8; ++j)
          fb.u[j] = *reinterpret_cast<const unsigned int*>(
              &sK[nt * 16 + ln][c * 32 + 2 * j + ((j >= 4) ? 8 : 0) + 8 * kh]);
        s = WMMA_F16(aq[c].v, fb.v, s);
      }
#pragma unroll
      for (int r = 0; r < 8; ++r) s[r] *= scale;
      sc[nt] = s;
    }

    // Dump scores C-layout -> LDS (fp32). Same-wave LDS ops are in-order.
#pragma unroll
    for (int nt = 0; nt < 2; ++nt)
#pragma unroll
      for (int r = 0; r < 8; ++r)
        sS[wave][r + 8 * kh][nt * 16 + ln] = sc[nt][r];

    // Row-per-lane softmax: lane (ln, kh) owns row ln, columns kh*16..+15.
    float4 rv[4];
    {
      const float4* rp =
          reinterpret_cast<const float4*>(&sS[wave][ln][kh * 16]);
#pragma unroll
      for (int q = 0; q < 4; ++q) rv[q] = rp[q];
    }
    float lmax = -3.0e38f;
#pragma unroll
    for (int q = 0; q < 4; ++q)
      lmax = fmaxf(lmax,
                   fmaxf(fmaxf(rv[q].x, rv[q].y), fmaxf(rv[q].z, rv[q].w)));
    lmax = fmaxf(lmax, __shfl_xor(lmax, 16, 32));  // full-row max
    const float newm = fmaxf(rm, lmax);
    const float f = __expf(rm - newm);
    rm = newm;
    float lsum = 0.f;
#pragma unroll
    for (int q = 0; q < 4; ++q) {
      const float p0 = __expf(rv[q].x - newm);
      const float p1 = __expf(rv[q].y - newm);
      const float p2 = __expf(rv[q].z - newm);
      const float p3 = __expf(rv[q].w - newm);
      lsum += (p0 + p1) + (p2 + p3);
      U32H w0; w0.h[0] = (_Float16)p0; w0.h[1] = (_Float16)p1;
      U32H w1; w1.h[0] = (_Float16)p2; w1.h[1] = (_Float16)p3;
      *reinterpret_cast<unsigned int*>(&sP[wave][ln][kh * 16 + q * 4 + 0]) =
          w0.u;
      *reinterpret_cast<unsigned int*>(&sP[wave][ln][kh * 16 + q * 4 + 2]) =
          w1.u;
    }
    lsum += __shfl_xor(lsum, 16, 32);  // full-row sum
    rs = rs * f + lsum;

    // Broadcast per-row rescale factor back to C layout; rescale O.
#pragma unroll
    for (int r = 0; r < 8; ++r) {
      const float fr = __shfl(f, r + 8 * kh, 32);
#pragma unroll
      for (int t = 0; t < 4; ++t) accO[t][r] *= fr;
    }

    // A-fragment of P from LDS.
    FragH ap;
#pragma unroll
    for (int j = 0; j < 8; ++j) {
      const int kA = 2 * j + ((j >= 4) ? 8 : 0) + 8 * kh;
      ap.u[j] = *reinterpret_cast<const unsigned int*>(&sP[wave][ln][kA]);
    }
    // O += P * V
#pragma unroll
    for (int t = 0; t < 4; ++t) {
      FragH fb;
#pragma unroll
      for (int j = 0; j < 8; ++j)
        fb.u[j] = *reinterpret_cast<const unsigned int*>(
            &sVt[t * 16 + ln][2 * j + ((j >= 4) ? 8 : 0) + 8 * kh]);
      accO[t] = WMMA_F16(ap.v, fb.v, accO[t]);
    }
  }

  // Normalize and store f16 head output (broadcast row sums to C layout).
#pragma unroll
  for (int r = 0; r < 8; ++r) {
    const float rsr = __shfl(rs, r + 8 * kh, 32);
    const float inv = 1.0f / rsr;
    const size_t row = (size_t)b * 1024 + q0 + wave * 16 + r + 8 * kh;
#pragma unroll
    for (int t = 0; t < 4; ++t)
      O[row * 768 + h * 64 + t * 16 + ln] = (_Float16)(accO[t][r] * inv);
  }
}

// ---------------------------------------------------------------------------
// Host-side orchestration
// ---------------------------------------------------------------------------
extern "C" void kernel_launch(void* const* d_in, const int* in_sizes, int n_in,
                              void* d_out, int out_size, void* d_ws,
                              size_t ws_size, hipStream_t stream) {
  (void)in_sizes; (void)n_in; (void)out_size; (void)ws_size;
  const float* x   = (const float*)d_in[0];
  const float* enc = (const float*)d_in[1];
  const float* Wq  = (const float*)d_in[2];
  const float* bq  = (const float*)d_in[3];
  const float* Wk  = (const float*)d_in[4];
  const float* bk  = (const float*)d_in[5];
  const float* Wv  = (const float*)d_in[6];
  const float* bv  = (const float*)d_in[7];
  const float* Wp  = (const float*)d_in[8];
  const float* bp  = (const float*)d_in[9];

  const int B = 8, NQ = 1024, E = 768, ENC = 1024;
  const int M = B * NQ;  // 8192
  const size_t nX   = (size_t)M * E;    // 6291456
  const size_t nEnc = (size_t)M * ENC;  // 8388608
  const size_t nWq  = (size_t)E * E;
  const size_t nWk  = (size_t)ENC * E;

  char* ws = (char*)d_ws;
  size_t off = 0;
  auto alloc = [&](size_t bytes) -> char* {
    char* p = ws + off;
    off += (bytes + 255) & ~(size_t)255;
    return p;
  };
  _Float16* x16  = (_Float16*)alloc(nX * 2);
  _Float16* e16  = (_Float16*)alloc(nEnc * 2);
  _Float16* wq16 = (_Float16*)alloc(nWq * 2);
  _Float16* wk16 = (_Float16*)alloc(nWk * 2);
  _Float16* wv16 = (_Float16*)alloc(nWk * 2);
  _Float16* wp16 = (_Float16*)alloc(nWq * 2);
  _Float16* Qb   = (_Float16*)alloc(nX * 2);
  _Float16* Kb   = (_Float16*)alloc(nX * 2);
  _Float16* Vb   = (_Float16*)alloc(nX * 2);
  _Float16* Ob   = (_Float16*)alloc(nX * 2);

  auto cvt = [&](const float* src, _Float16* dst, size_t n) {
    int blocks = (int)((n + 255) / 256);
    if (blocks > 4096) blocks = 4096;
    cvt_f32_to_f16<<<blocks, 256, 0, stream>>>(src, dst, (int)n);
  };
  cvt(x, x16, nX);
  cvt(enc, e16, nEnc);
  cvt(Wq, wq16, nWq);
  cvt(Wk, wk16, nWk);
  cvt(Wv, wv16, nWk);
  cvt(Wp, wp16, nWq);

  // Projections: Q = x*Wq+bq ; K = enc*Wk+bk ; V = enc*Wv+bv  (f16 out)
  dim3 gemmGrid(M / 64, E / 64);
  gemm_wmma<false><<<gemmGrid, 128, 0, stream>>>(x16, wq16, bq, Qb, M, E, E);
  gemm_wmma<false><<<gemmGrid, 128, 0, stream>>>(e16, wk16, bk, Kb, M, ENC, E);
  gemm_wmma<false><<<gemmGrid, 128, 0, stream>>>(e16, wv16, bv, Vb, M, ENC, E);

  // Flash attention per (b, h, 64-row q block).
  dim3 attnGrid(NQ / 64, 12, B);
  attn_wmma<<<attnGrid, 128, 0, stream>>>(Qb, Kb, Vb, Ob);

  // Output projection: out = O*Wp + bp  (fp32 out -> d_out)
  gemm_wmma<true><<<gemmGrid, 128, 0, stream>>>(Ob, wp16, bp, d_out, M, E, E);
}
